// Transformer_16028817949203
// MI455X (gfx1250) — compile-verified, matrix operand order checked by reading, not run on hardware
//
#include <hip/hip_runtime.h>
#include <hip/hip_bf16.h>

// ---------------------------------------------------------------------------
// MI455X (gfx1250) transformer forward: bf16 WMMA GEMMs + flash attention.
// GEMM A-tiles are moved by the Tensor Data Mover (tensor_load_to_lds,
// TENSORcnt-synchronized), B-tiles are register-staged (f32->bf16 convert).
// B=4, N=2048, DIM=512, DEPTH=4, NH=8, DH=64, MLP=2048. fp32 reference I/O.
// ---------------------------------------------------------------------------

typedef __bf16        v16bf __attribute__((ext_vector_type(16)));
typedef float         v8f   __attribute__((ext_vector_type(8)));
typedef unsigned int  v8u   __attribute__((ext_vector_type(8)));
typedef unsigned int  u32x4 __attribute__((ext_vector_type(4)));
typedef int           i32x8 __attribute__((ext_vector_type(8)));
typedef int           i32x4 __attribute__((ext_vector_type(4)));

union Frag { v16bf v; v8u u; };

#define TOK    8192          // 4 * 2048 tokens
#define SEQ    2048
#define DIM    512
#define NHEAD  8
#define DH     64
#define QKVLD  1536          // 3 * NH * DH
#define MLPD   2048
#define ATTN_SCALE 0.125f    // 64^-0.5

static __device__ __forceinline__ unsigned short f2bf(float x) {
  union { float f; unsigned int u; } c; c.f = x;
  unsigned int r = c.u + 0x7FFFu + ((c.u >> 16) & 1u);   // round-nearest-even
  return (unsigned short)(r >> 16);
}

// K-pair offset inside a 32-wide K chunk for 16-bit A/B WMMA fragments:
// VGPR r, lane group g (0: lanes 0-15, 1: lanes 16-31).
static __device__ __forceinline__ int kpair(int r, int g) {
  return ((r < 4) ? (2 * r) : (2 * r + 8)) + 8 * g;
}

// ---------------------------------------------------------------------------
// TDM: DMA a 64-row x 32-col bf16 tile (row stride = K elements) from global
// into LDS at lds_off, padding +1 DWORD after every 16 DWORDs (row stride 34
// ushorts -> matches As[64][34], conflict-free fragment reads).
// D# layout per CDNA5 ISA ch.8 (2-D tile; groups 2/3 zero).
// This toolchain's builtin takes 6 args (extra int32x8 before cpol).
// ---------------------------------------------------------------------------
static __device__ __forceinline__ void tdm_load_a_tile(
    const unsigned short* gptr, unsigned lds_off, int K) {
  const unsigned long long ga = (unsigned long long)(size_t)gptr;
  u32x4 g0;
  g0[0] = 1u;                                              // count=1, user mode
  g0[1] = lds_off;                                         // lds_addr (bytes)
  g0[2] = (unsigned)(ga & 0xFFFFFFFFull);                  // global_addr[31:0]
  g0[3] = (unsigned)((ga >> 32) & 0x01FFFFFFull)           // global_addr[56:32]
          | (2u << 30);                                    // type=2 (image)
  i32x8 g1;
  g1[0] = (int)((1u << 16)      // data_size = 1 -> 2 bytes
              | (1u << 20)      // pad_enable
              | (3u << 22));    // pad_interval = 16 DWORDs; pad_amount=0 -> 1 DWORD
  g1[1] = (int)(((unsigned)K & 0xFFFFu) << 16);            // tensor_dim0 lo16
  g1[2] = (int)((((unsigned)K >> 16) & 0xFFFFu)            // tensor_dim0 hi16
              | (0xFFFFu << 16));                          // tensor_dim1 lo16 (big)
  g1[3] = (int)(32u << 16);                                // dim1 hi=0 | tile_dim0=32
  g1[4] = (int)64u;                                        // tile_dim1=64, tile_dim2=0
  g1[5] = (int)(unsigned)K;                                // tensor_dim0_stride lo32
  g1[6] = 0;                                               // stride hi / dim1_stride
  g1[7] = 0;
  const i32x4 gz4 = {0, 0, 0, 0};
  const i32x8 gz8 = {0, 0, 0, 0, 0, 0, 0, 0};
  __builtin_amdgcn_tensor_load_to_lds(g0, g1, gz4, gz4, gz8, 0);
}

// ---------------------------------------------------------------------------
// LayerNorm over last dim (512). One block (128 threads) per token row.
// ---------------------------------------------------------------------------
__global__ void ln_kernel(const float* __restrict__ x,
                          const float* __restrict__ gamma,
                          const float* __restrict__ beta,
                          unsigned short* __restrict__ outb,
                          float* __restrict__ outf) {
  __shared__ float red[128];
  const int row = blockIdx.x;
  const int tid = threadIdx.x;
  const float* xr = x + (size_t)row * DIM;

  float v[4];
  float s = 0.f;
#pragma unroll
  for (int t = 0; t < 4; ++t) { v[t] = xr[tid + 128 * t]; s += v[t]; }
  red[tid] = s; __syncthreads();
  for (int w = 64; w > 0; w >>= 1) { if (tid < w) red[tid] += red[tid + w]; __syncthreads(); }
  const float mu = red[0] * (1.f / DIM);
  __syncthreads();

  float sq = 0.f;
#pragma unroll
  for (int t = 0; t < 4; ++t) { float d = v[t] - mu; sq += d * d; }
  red[tid] = sq; __syncthreads();
  for (int w = 64; w > 0; w >>= 1) { if (tid < w) red[tid] += red[tid + w]; __syncthreads(); }
  const float rstd = rsqrtf(red[0] * (1.f / DIM) + 1e-5f);

#pragma unroll
  for (int t = 0; t < 4; ++t) {
    const int col = tid + 128 * t;
    const float y = (v[t] - mu) * rstd * gamma[col] + beta[col];
    if (outb) outb[(size_t)row * DIM + col] = f2bf(y);
    if (outf) outf[(size_t)row * DIM + col] = y;
  }
}

// ---------------------------------------------------------------------------
// Tiled WMMA GEMM: C[M,N] = A[M,K](bf16) * B[K,N](f32->bf16) + epilogue.
// Block tile 64x128, 8 waves (256 threads), each wave 32x32 (2x2 WMMA tiles).
// Pipelined + double-buffered LDS: TDM DMAs A tile kt+1 while B tile kt+1 is
// register-staged and WMMAs consume tile kt. Wave 0 waits TENSORcnt, the
// barrier publishes the tile to all waves.
// ---------------------------------------------------------------------------
template <bool HAS_BIAS, bool DO_GELU, bool ADD_RES, bool OUT_F32>
__global__ void gemm_wmma(const unsigned short* __restrict__ A,
                          const float* __restrict__ B,
                          const float* __restrict__ bias,
                          const float* __restrict__ res,
                          float* __restrict__ outf,
                          unsigned short* __restrict__ outb,
                          int M, int N, int K) {
  __shared__ unsigned short As[2][64][34];    // [buf][m][k], TDM-padded
  __shared__ unsigned short Bs[2][128][34];   // [buf][n][k] (transposed), padded

  const int tid  = threadIdx.x;            // 0..255
  const int wave = tid >> 5;               // 0..7
  const int lane = tid & 31;
  const int g    = lane >> 4;
  const int l16  = lane & 15;
  const int wr   = wave >> 2;              // wave row (0..1)
  const int wc   = wave & 3;               // wave col (0..3)
  const int m0   = blockIdx.y * 64;
  const int n0   = blockIdx.x * 128;

  v8f acc[2][2];
#pragma unroll
  for (int i = 0; i < 2; ++i)
#pragma unroll
    for (int j = 0; j < 2; ++j)
#pragma unroll
      for (int r = 0; r < 8; ++r) acc[i][j][r] = 0.f;

  // Per-thread staging registers for one B K-chunk.
  const int bRow = tid >> 7, bCol = tid & 127;   // B: 2 k-rows apart
  float bR[16];

  const int KT = K >> 5;
  // Prologue: TDM A tile 0, stage B tile 0.
  if (tid == 0)
    tdm_load_a_tile(A + (size_t)m0 * K, (unsigned)(size_t)&As[0][0][0], K);
  {
    const float* Bp = B + (size_t)n0;
#pragma unroll
    for (int it = 0; it < 16; ++it)
      bR[it] = Bp[(size_t)(bRow + 2 * it) * N + bCol];
  }

  int buf = 0;
  for (int kt = 0; kt < KT; ++kt) {
    // Commit staged B tile to LDS[buf].
#pragma unroll
    for (int it = 0; it < 16; ++it)
      Bs[buf][bCol][bRow + 2 * it] = f2bf(bR[it]);
    // Wave 0: A tile[buf] DMA complete before the barrier releases anyone.
    if (tid < 32) __builtin_amdgcn_s_wait_tensorcnt(0);
    __syncthreads();

    // Kick next tile: TDM A into As[buf^1], stage next B into registers.
    if (kt + 1 < KT) {
      if (tid == 0)
        tdm_load_a_tile(A + (size_t)m0 * K + (kt + 1) * 32,
                        (unsigned)(size_t)&As[buf ^ 1][0][0], K);
      const float* Bp = B + (size_t)(kt + 1) * 32 * N + n0;
#pragma unroll
      for (int it = 0; it < 16; ++it)
        bR[it] = Bp[(size_t)(bRow + 2 * it) * N + bCol];
    }

    // All fragments first, then 4 back-to-back WMMAs (fills hazard slots).
    Frag a[2], b[2];
#pragma unroll
    for (int i = 0; i < 2; ++i) {
      const int m = wr * 32 + i * 16 + l16;
#pragma unroll
      for (int r = 0; r < 8; ++r)
        a[i].u[r] = *(const unsigned int*)&As[buf][m][kpair(r, g)];
    }
#pragma unroll
    for (int j = 0; j < 2; ++j) {
      const int n = wc * 32 + j * 16 + l16;
#pragma unroll
      for (int r = 0; r < 8; ++r)
        b[j].u[r] = *(const unsigned int*)&Bs[buf][n][kpair(r, g)];
    }
#pragma unroll
    for (int i = 0; i < 2; ++i)
#pragma unroll
      for (int j = 0; j < 2; ++j)
        acc[i][j] = __builtin_amdgcn_wmma_f32_16x16x32_bf16(
            false, a[i].v, false, b[j].v, (short)0, acc[i][j], false, false);
    buf ^= 1;
  }

  // Epilogue. C layout: lane holds col n=l16, rows m = r + 8*g.
#pragma unroll
  for (int i = 0; i < 2; ++i)
#pragma unroll
    for (int j = 0; j < 2; ++j)
#pragma unroll
      for (int r = 0; r < 8; ++r) {
        const int m = m0 + wr * 32 + i * 16 + g * 8 + r;
        const int n = n0 + wc * 32 + j * 16 + l16;
        float v = acc[i][j][r];
        if constexpr (HAS_BIAS) v += bias[n];
        if constexpr (DO_GELU)  v = 0.5f * v * (1.f + erff(v * 0.70710678118654752f));
        const size_t idx = (size_t)m * N + n;
        if constexpr (ADD_RES)  v += res[idx];
        if constexpr (OUT_F32)  outf[idx] = v;
        else                    outb[idx] = f2bf(v);
      }
}

// ---------------------------------------------------------------------------
// Flash attention. Grid: (SEQ/64, B*NH). Block 128 threads = 4 waves; each
// wave owns 16 query rows. Q staged in LDS once; all WMMA operand fragments
// are gathered into registers before each 4-WMMA burst.
// qkv: [TOK][1536] bf16 (q | k | v, each NH*DH). out: [TOK][512] bf16.
// ---------------------------------------------------------------------------
__global__ void attn_kernel(const unsigned short* __restrict__ qkv,
                            unsigned short* __restrict__ out) {
  __shared__ unsigned short Ql[64][66];      // [qrow][dh]
  __shared__ unsigned short Kl[64][66];      // [key][dh]
  __shared__ unsigned short Vt[64][66];      // [dh][key]
  __shared__ unsigned short Pl[4][16][66];   // per-wave P staging [m][key]

  const int tid   = threadIdx.x;
  const int wave  = tid >> 5;
  const int lane  = tid & 31;
  const int g     = lane >> 4;
  const int l16   = lane & 15;
  const int bh    = blockIdx.y;
  const int batch = bh >> 3;
  const int head  = bh & 7;
  const int tok0  = batch * SEQ + blockIdx.x * 64;
  const int qoff  = head * DH;

  // Stage Q tile (64 rows x 64 dh) into LDS once; reused for all key tiles.
  for (int i = tid; i < 64 * 64; i += 128) {
    const int qr = i >> 6, d = i & 63;
    Ql[qr][d] = qkv[(size_t)(tok0 + qr) * QKVLD + qoff + d];
  }

  v8f o[4];
  float mstat[8], lstat[8];
#pragma unroll
  for (int t = 0; t < 4; ++t)
#pragma unroll
    for (int r = 0; r < 8; ++r) o[t][r] = 0.f;
#pragma unroll
  for (int r = 0; r < 8; ++r) { mstat[r] = -1e30f; lstat[r] = 0.f; }

  for (int j = 0; j < SEQ; j += 64) {
    __syncthreads();
    const int kb = batch * SEQ + j;
    for (int i = tid; i < 64 * 64; i += 128) {
      const int key = i >> 6, d = i & 63;
      Kl[key][d] = qkv[(size_t)(kb + key) * QKVLD + DIM + qoff + d];
    }
    for (int i = tid; i < 64 * 64; i += 128) {
      const int key = i >> 6, d = i & 63;
      Vt[d][key] = qkv[(size_t)(kb + key) * QKVLD + 2 * DIM + qoff + d];
    }
    // Prefetch next tile's K/V rows (global_prefetch_b8).
    if (j + 64 < SEQ) {
      const int nk = kb + 64 + (tid & 63);
      if (tid < 64)       __builtin_prefetch(&qkv[(size_t)nk * QKVLD + DIM + qoff], 0, 0);
      else                __builtin_prefetch(&qkv[(size_t)nk * QKVLD + 2 * DIM + qoff], 0, 0);
    }
    __syncthreads();

    // S = Q @ K^T : 2 dh-chunks; per chunk load qa + all 4 K fragments,
    // then 4 independent WMMAs back-to-back.
    v8f s[4];
#pragma unroll
    for (int t = 0; t < 4; ++t)
#pragma unroll
      for (int r = 0; r < 8; ++r) s[t][r] = 0.f;
#pragma unroll
    for (int c = 0; c < 2; ++c) {
      Frag qa, kbf[4];
#pragma unroll
      for (int r = 0; r < 8; ++r)
        qa.u[r] = *(const unsigned int*)&Ql[wave * 16 + l16][c * 32 + kpair(r, g)];
#pragma unroll
      for (int t = 0; t < 4; ++t)
#pragma unroll
        for (int r = 0; r < 8; ++r)
          kbf[t].u[r] = *(const unsigned int*)&Kl[t * 16 + l16][c * 32 + kpair(r, g)];
#pragma unroll
      for (int t = 0; t < 4; ++t)
        s[t] = __builtin_amdgcn_wmma_f32_16x16x32_bf16(
            false, qa.v, false, kbf[t].v, (short)0, s[t], false, false);
    }

    // Online softmax (rows m = r + 8g live across the 16 lanes of a group).
#pragma unroll
    for (int r = 0; r < 8; ++r) {
      float mx = -1e30f;
#pragma unroll
      for (int t = 0; t < 4; ++t) mx = fmaxf(mx, s[t][r]);
      for (int d = 1; d < 16; d <<= 1) mx = fmaxf(mx, __shfl_xor(mx, d, 32));
      mx *= ATTN_SCALE;
      const float mnew  = fmaxf(mstat[r], mx);
      const float alpha = __expf(mstat[r] - mnew);
      mstat[r] = mnew;
      float rs = 0.f;
#pragma unroll
      for (int t = 0; t < 4; ++t) {
        const float p = __expf(s[t][r] * ATTN_SCALE - mnew);
        s[t][r] = p;
        rs += p;
      }
      for (int d = 1; d < 16; d <<= 1) rs += __shfl_xor(rs, d, 32);
      lstat[r] = lstat[r] * alpha + rs;
#pragma unroll
      for (int t = 0; t < 4; ++t) o[t][r] *= alpha;
    }

    // Re-shape P: C-layout -> LDS -> A-layout (per-wave region, in-wave dep).
#pragma unroll
    for (int t = 0; t < 4; ++t)
#pragma unroll
      for (int r = 0; r < 8; ++r)
        Pl[wave][g * 8 + r][t * 16 + l16] = f2bf(s[t][r]);

    // O += P @ V : 2 key-chunks; load pf + all 4 V fragments, then 4 WMMAs.
#pragma unroll
    for (int c = 0; c < 2; ++c) {
      Frag pf, vb[4];
#pragma unroll
      for (int r = 0; r < 8; ++r)
        pf.u[r] = *(const unsigned int*)&Pl[wave][l16][c * 32 + kpair(r, g)];
#pragma unroll
      for (int t = 0; t < 4; ++t)
#pragma unroll
        for (int r = 0; r < 8; ++r)
          vb[t].u[r] = *(const unsigned int*)&Vt[t * 16 + l16][c * 32 + kpair(r, g)];
#pragma unroll
      for (int t = 0; t < 4; ++t)
        o[t] = __builtin_amdgcn_wmma_f32_16x16x32_bf16(
            false, pf.v, false, vb[t].v, (short)0, o[t], false, false);
    }
  }

  // Normalize and store (merge heads back into [TOK][512]).
#pragma unroll
  for (int t = 0; t < 4; ++t)
#pragma unroll
    for (int r = 0; r < 8; ++r) {
      const int m = g * 8 + r;
      const float v = o[t][r] / lstat[r];
      out[(size_t)(tok0 + wave * 16 + m) * DIM + qoff + t * 16 + l16] = f2bf(v);
    }
}

// ---------------------------------------------------------------------------
// Host orchestration.
// ---------------------------------------------------------------------------
extern "C" void kernel_launch(void* const* d_in, const int* in_sizes, int n_in,
                              void* d_out, int out_size, void* d_ws, size_t ws_size,
                              hipStream_t stream) {
  (void)in_sizes; (void)n_in; (void)out_size; (void)ws_size;
  const float* x    = (const float*)d_in[0];
  const float* Wqkv = (const float*)d_in[1];
  const float* Wo   = (const float*)d_in[2];
  const float* bo   = (const float*)d_in[3];
  const float* ln1g = (const float*)d_in[4];
  const float* ln1b = (const float*)d_in[5];
  const float* W1   = (const float*)d_in[6];
  const float* b1   = (const float*)d_in[7];
  const float* W2   = (const float*)d_in[8];
  const float* b2   = (const float*)d_in[9];
  const float* ln2g = (const float*)d_in[10];
  const float* ln2b = (const float*)d_in[11];
  const float* lnfg = (const float*)d_in[12];
  const float* lnfb = (const float*)d_in[13];

  char* ws = (char*)d_ws;
  float*          xf     = (float*)ws;          ws += (size_t)TOK * DIM * 4;
  unsigned short* hbf    = (unsigned short*)ws; ws += (size_t)TOK * DIM * 2;
  unsigned short* qkvbf  = (unsigned short*)ws; ws += (size_t)TOK * QKVLD * 2;
  unsigned short* attnbf = (unsigned short*)ws; ws += (size_t)TOK * DIM * 2;
  unsigned short* mlpbf  = (unsigned short*)ws; ws += (size_t)TOK * MLPD * 2;

  (void)hipMemcpyAsync(xf, x, (size_t)TOK * DIM * 4, hipMemcpyDeviceToDevice, stream);

  const dim3 blk128(128);
  const dim3 blk256(256);
  for (int l = 0; l < 4; ++l) {
    // PreNorm 1 -> bf16
    ln_kernel<<<TOK, blk128, 0, stream>>>(xf, ln1g + l * DIM, ln1b + l * DIM, hbf, nullptr);
    // QKV projection (no bias) -> bf16
    gemm_wmma<false, false, false, false><<<dim3(QKVLD / 128, TOK / 64), blk256, 0, stream>>>(
        hbf, Wqkv + (size_t)l * DIM * QKVLD, nullptr, nullptr, nullptr, qkvbf,
        TOK, QKVLD, DIM);
    // Attention
    attn_kernel<<<dim3(SEQ / 64, 4 * NHEAD), blk128, 0, stream>>>(qkvbf, attnbf);
    // Output projection + bias + residual -> f32 residual stream
    gemm_wmma<true, false, true, true><<<dim3(DIM / 128, TOK / 64), blk256, 0, stream>>>(
        attnbf, Wo + (size_t)l * DIM * DIM, bo + l * DIM, xf, xf, nullptr,
        TOK, DIM, DIM);
    // PreNorm 2 -> bf16
    ln_kernel<<<TOK, blk128, 0, stream>>>(xf, ln2g + l * DIM, ln2b + l * DIM, hbf, nullptr);
    // MLP up + bias + exact GELU -> bf16
    gemm_wmma<true, true, false, false><<<dim3(MLPD / 128, TOK / 64), blk256, 0, stream>>>(
        hbf, W1 + (size_t)l * DIM * MLPD, b1 + l * MLPD, nullptr, nullptr, mlpbf,
        TOK, MLPD, DIM);
    // MLP down + bias + residual -> f32 residual stream
    gemm_wmma<true, false, true, true><<<dim3(DIM / 128, TOK / 64), blk256, 0, stream>>>(
        mlpbf, W2 + (size_t)l * MLPD * DIM, b2 + l * DIM, xf, xf, nullptr,
        TOK, DIM, MLPD);
  }
  // Final LayerNorm -> f32 output
  ln_kernel<<<TOK, blk128, 0, stream>>>(xf, lnfg, lnfb, nullptr, (float*)d_out);
}
